// Transformer_20392504722107
// MI455X (gfx1250) — matrix operand order repaired and checked by reading, compiled, not run on hardware
//
#include <hip/hip_runtime.h>

typedef __attribute__((ext_vector_type(16))) __bf16 v16bf;
typedef __attribute__((ext_vector_type(8)))  __bf16 v8bf;
typedef __attribute__((ext_vector_type(4)))  __bf16 v4bf;
typedef __attribute__((ext_vector_type(8)))  float  v8f;
typedef __attribute__((ext_vector_type(4)))  float  v4f;
typedef __attribute__((ext_vector_type(4)))  int    v4i;

#define BSZ 512
#define NE  128
#define FD  512
#define NH  8
#define HD  64
#define LDA 520   // padded LDS row stride (elems) for staged activations

// ---- workspace layout (bytes) ----
static constexpr size_t SZ_ACT  = (size_t)BSZ * NE * FD * 2;          // 64 MiB bf16
static constexpr size_t OFF_INP = 0;                                   // inp, later att
static constexpr size_t OFF_Q   = OFF_INP + SZ_ACT;
static constexpr size_t OFF_K   = OFF_Q   + SZ_ACT;
static constexpr size_t OFF_V   = OFF_K   + SZ_ACT;
static constexpr size_t OFF_WQKT= OFF_V   + SZ_ACT;                    // 1024x512 bf16
static constexpr size_t OFF_WVT = OFF_WQKT + (size_t)1024 * 512 * 2;   // 512x512 bf16
static constexpr size_t OFF_WMT = OFF_WVT  + (size_t)512  * 512 * 2;   // 512x512 bf16
static constexpr size_t OFF_Y   = OFF_Q;                               // f32 y_pre aliases q+k
static constexpr size_t OFF_ATT = OFF_INP;

static constexpr size_t SMEM_ACT = (size_t)NE * LDA * 2;               // 133,120 B dynamic LDS

// ---- helpers ----
__device__ __forceinline__ v8f wmma_bf16(v16bf a, v16bf b, v8f c) {
  return __builtin_amdgcn_wmma_f32_16x16x32_bf16(false, a, false, b, (short)0, c, false, false);
}

// A operand (16x32 bf16): lane l holds row M = mrow + (l&15),
// K = k0 + kh*16 + {0..15} (contiguous, matches B)
__device__ __forceinline__ v16bf load_a_tile(const __bf16* base, int ld, int mrow,
                                             int k0, int n, int kh) {
  const __bf16* p = base + (size_t)(mrow + n) * ld + k0 + kh * 16;
  v8bf lo = *(const v8bf*)p;
  v8bf hi = *(const v8bf*)(p + 8);
  return __builtin_shufflevector(lo, hi, 0,1,2,3,4,5,6,7,8,9,10,11,12,13,14,15);
}

// 16-byte async copy global -> LDS (ASYNCcnt-tracked, no VGPR data path)
__device__ __forceinline__ void async_copy16(const __bf16* g, __bf16* l) {
  __builtin_amdgcn_global_load_async_to_lds_b128(
      (__attribute__((address_space(1))) v4i*)(void*)g,
      (__attribute__((address_space(3))) v4i*)(void*)l, 0, 0);
}
__device__ __forceinline__ void wait_async_all() {
  asm volatile("s_wait_asynccnt 0" ::: "memory");
}
// Block LICM from hoisting/caching LDS reads across outer job loops
// (prevents the compiler from materializing all A-tiles and spilling to scratch).
__device__ __forceinline__ void no_hoist() {
  asm volatile("" ::: "memory");
}

// Stage a 128x512 bf16 activation tile (row-major, global) into LDS with LDA stride.
__device__ __forceinline__ void stage_act_async(const __bf16* g, __bf16* s, int tid) {
#pragma unroll
  for (int it = 0; it < 32; ++it) {
    int chunk = it * 256 + tid;          // 0..8191, 8 elems (16B) each
    int row = chunk >> 6, cc = chunk & 63;
    async_copy16(g + ((size_t)row << 9) + cc * 8, s + row * LDA + cc * 8);
  }
  wait_async_all();
  __syncthreads();
}

__device__ __forceinline__ float waveRedSum(float v) {
#pragma unroll
  for (int m = 16; m > 0; m >>= 1) v += __shfl_xor(v, m, 32);
  return v;
}
__device__ __forceinline__ float halfRedSum(float v) {   // within 16-lane groups
#pragma unroll
  for (int m = 8; m > 0; m >>= 1) v += __shfl_xor(v, m, 32);
  return v;
}
__device__ __forceinline__ float halfRedMax(float v) {
#pragma unroll
  for (int m = 8; m > 0; m >>= 1) v = fmaxf(v, __shfl_xor(v, m, 32));
  return v;
}

// ---- K0a: transpose + convert f32 weight (FD x C) -> bf16 (C x FD) ----
__global__ void __launch_bounds__(256)
k_transpose_cvt(const float* __restrict__ src, __bf16* __restrict__ dst, int C) {
  int idx = blockIdx.x * 256 + threadIdx.x;
  if (idx >= C * FD) return;
  int c = idx >> 9, f = idx & (FD - 1);
  dst[idx] = (__bf16)src[(size_t)f * C + c];
}

// ---- K0b: LayerNorm-pre: x f32 -> inp bf16 (one wave per token row of 512) ----
__global__ void __launch_bounds__(256)
k_ln_pre(const float* __restrict__ x, const float* __restrict__ g,
         const float* __restrict__ bp, __bf16* __restrict__ inp) {
  int wid = threadIdx.x >> 5, lane = threadIdx.x & 31;
  size_t row = (size_t)blockIdx.x * 8 + wid;          // 0 .. BSZ*NE-1
  const float* xr = x + row * FD;
  v4f u[4];
#pragma unroll
  for (int j = 0; j < 4; ++j) u[j] = *(const v4f*)(xr + j * 128 + lane * 4);
  float s = 0.f;
#pragma unroll
  for (int j = 0; j < 4; ++j)
#pragma unroll
    for (int i = 0; i < 4; ++i) s += u[j][i];
  float mu = waveRedSum(s) * (1.f / FD);
  float vs = 0.f;
#pragma unroll
  for (int j = 0; j < 4; ++j)
#pragma unroll
    for (int i = 0; i < 4; ++i) { float d = u[j][i] - mu; vs += d * d; }
  float rs = rsqrtf(waveRedSum(vs) * (1.f / FD) + 1e-3f);
  __bf16* o = inp + row * FD;
#pragma unroll
  for (int j = 0; j < 4; ++j) {
    int c0 = j * 128 + lane * 4;
    v4f g4 = *(const v4f*)(g + c0);
    v4f b4 = *(const v4f*)(bp + c0);
    v4bf ob;
#pragma unroll
    for (int i = 0; i < 4; ++i) ob[i] = (__bf16)((u[j][i] - mu) * rs * g4[i] + b4[i]);
    *(v4bf*)(o + c0) = ob;
  }
}

// ---- K1: QKV projection per batch. q,k,v row-major [b][h][n][64] bf16 ----
// A-operand staged once into LDS via async copies; B streams from transposed weights.
__global__ void __launch_bounds__(256)
k_qkv(const __bf16* __restrict__ inp, const __bf16* __restrict__ wqkT,
      const __bf16* __restrict__ wvT, const float* __restrict__ bqk,
      const float* __restrict__ bv, __bf16* __restrict__ q,
      __bf16* __restrict__ k, __bf16* __restrict__ v) {
  extern __shared__ __bf16 s_a[];          // NE x LDA
  int b = blockIdx.x;
  int tid = threadIdx.x;
  int wid = tid >> 5, lane = tid & 31;
  int n = lane & 15, kh = lane >> 4;

  stage_act_async(inp + (size_t)b * NE * FD, s_a, tid);

  for (int t = 0; t < 12; ++t) {
    no_hoist();                     // force per-job ds_load of A tiles (no spill)
    int job = wid + t * 8;          // 0..95
    int head = job / 12, s = job % 12;
    int kind = s >> 2;              // 0=q 1=k 2=v
    int nt = s & 3;
    int cl = nt * 16 + n;           // column within head (this lane)
    const __bf16* WT; float bias; __bf16* out;
    if (kind == 0)      { int c = 2 * (head * HD + cl);     WT = wqkT + (size_t)c * FD; bias = bqk[c]; out = q; }
    else if (kind == 1) { int c = 2 * (head * HD + cl) + 1; WT = wqkT + (size_t)c * FD; bias = bqk[c]; out = k; }
    else                { int c = head * HD + cl;           WT = wvT  + (size_t)c * FD; bias = bv[c];  out = v; }
    out += (size_t)(b * NH + head) * NE * HD;

    v8f acc[8];
#pragma unroll
    for (int m = 0; m < 8; ++m)
#pragma unroll
      for (int r = 0; r < 8; ++r) acc[m][r] = bias;

    for (int ks = 0; ks < 16; ++ks) {
      v16bf bm = *(const v16bf*)(WT + ks * 32 + kh * 16);
#pragma unroll
      for (int m = 0; m < 8; ++m) {
        v16bf am = load_a_tile(s_a, LDA, m * 16, ks * 32, n, kh);
        acc[m] = wmma_bf16(am, bm, acc[m]);
      }
    }
#pragma unroll
    for (int m = 0; m < 8; ++m)
#pragma unroll
      for (int r = 0; r < 8; ++r)
        out[(size_t)(m * 16 + r + 8 * kh) * HD + nt * 16 + n] = (__bf16)acc[m][r];
  }
}

// ---- K2: attention per (head, batch). att row-major [b][n][h*64+d] bf16 ----
__global__ void __launch_bounds__(256)
k_attn(const __bf16* __restrict__ q, const __bf16* __restrict__ k,
       const __bf16* __restrict__ v, const float* __restrict__ mask,
       __bf16* __restrict__ att) {
  __shared__ __bf16 s_p[NE * 136];    // p row-major, padded stride
  __shared__ __bf16 s_vT[HD * 144];   // v transposed [d][m], padded stride
  __shared__ float  s_mask[NE];

  int h = blockIdx.x, b = blockIdx.y;
  int tid = threadIdx.x, wid = tid >> 5, lane = tid & 31;
  int n = lane & 15, kh = lane >> 4;
  size_t hb = (size_t)(b * NH + h) * NE * HD;
  const __bf16* qh = q + hb;
  const __bf16* kk = k + hb;
  const __bf16* vh = v + hb;

  if (tid < NE) s_mask[tid] = mask[(size_t)b * NE + tid];
  // stage v transposed into LDS
#pragma unroll
  for (int c2 = 0; c2 < 4; ++c2) {
    int lin = (c2 * 256 + tid) * 8;   // element index in 128x64
    int m = lin >> 6, d0 = lin & 63;
    v8bf vv = *(const v8bf*)(vh + lin);
#pragma unroll
    for (int i = 0; i < 8; ++i) s_vT[(d0 + i) * 144 + m] = vv[i];
  }
  __syncthreads();

  // logits = q @ k^T : M = 16 queries (tile wid), N = 128 keys, K = 64
  v8f L[8] = {};
#pragma unroll
  for (int ks = 0; ks < 2; ++ks) {
    v16bf am = load_a_tile(qh, HD, wid * 16, ks * 32, n, kh);
#pragma unroll
    for (int nt = 0; nt < 8; ++nt) {
      v16bf bm = *(const v16bf*)(kk + (size_t)(nt * 16 + n) * HD + ks * 32 + kh * 16);
      L[nt] = wmma_bf16(am, bm, L[nt]);
    }
  }
  // scale + mask
#pragma unroll
  for (int nt = 0; nt < 8; ++nt) {
    float mc = s_mask[nt * 16 + n];
    float pen = (1.f - mc) * 1e10f;
#pragma unroll
    for (int r = 0; r < 8; ++r) L[nt][r] = L[nt][r] * 0.125f - pen;
  }
  // row softmax (rows live in 16-lane groups: M = 16*wid + r + 8*kh)
  float rmax[8];
#pragma unroll
  for (int r = 0; r < 8; ++r) {
    float mx = -3.4e38f;
#pragma unroll
    for (int nt = 0; nt < 8; ++nt) mx = fmaxf(mx, L[nt][r]);
    rmax[r] = halfRedMax(mx);
  }
#pragma unroll
  for (int nt = 0; nt < 8; ++nt) {
    float mc = s_mask[nt * 16 + n];
#pragma unroll
    for (int r = 0; r < 8; ++r) L[nt][r] = __expf(L[nt][r] - rmax[r]) * mc;
  }
  float rinv[8];
#pragma unroll
  for (int r = 0; r < 8; ++r) {
    float ssum = 0.f;
#pragma unroll
    for (int nt = 0; nt < 8; ++nt) ssum += L[nt][r];
    rinv[r] = 1.f / (halfRedSum(ssum) + 1e-10f);
  }
  // store p (C-layout -> row-major LDS)
#pragma unroll
  for (int nt = 0; nt < 8; ++nt)
#pragma unroll
    for (int r = 0; r < 8; ++r)
      s_p[(wid * 16 + r + 8 * kh) * 136 + nt * 16 + n] = (__bf16)(L[nt][r] * rinv[r]);
  __syncthreads();

  // att = p @ v : M = 16 queries, N = 64, K = 128
  v8f Oc[4] = {};
#pragma unroll
  for (int ks = 0; ks < 4; ++ks) {
    const __bf16* pa = &s_p[(wid * 16 + n) * 136 + ks * 32 + kh * 16];
    v8bf lo = *(const v8bf*)pa;
    v8bf hi = *(const v8bf*)(pa + 8);
    v16bf am = __builtin_shufflevector(lo, hi, 0,1,2,3,4,5,6,7,8,9,10,11,12,13,14,15);
#pragma unroll
    for (int nt = 0; nt < 4; ++nt) {
      v16bf bm = *(const v16bf*)(&s_vT[(nt * 16 + n) * 144 + ks * 32 + kh * 16]);
      Oc[nt] = wmma_bf16(am, bm, Oc[nt]);
    }
  }
  __bf16* ob = att + (size_t)b * NE * FD;
#pragma unroll
  for (int nt = 0; nt < 4; ++nt)
#pragma unroll
    for (int r = 0; r < 8; ++r)
      ob[(size_t)(wid * 16 + r + 8 * kh) * FD + h * HD + nt * 16 + n] = (__bf16)Oc[nt][r];
}

// ---- K3: y_pre = att @ Wmlp + bmlp + x  (f32 out), att staged in LDS ----
__global__ void __launch_bounds__(256)
k_outproj(const __bf16* __restrict__ att, const __bf16* __restrict__ wmT,
          const float* __restrict__ bmlp, const float* __restrict__ x,
          float* __restrict__ y) {
  extern __shared__ __bf16 s_a[];      // NE x LDA
  int b = blockIdx.x;
  int tid = threadIdx.x;
  int wid = tid >> 5, lane = tid & 31;
  int n = lane & 15, kh = lane >> 4;
  const float* xr = x + (size_t)b * NE * FD;
  float* yr = y + (size_t)b * NE * FD;

  stage_act_async(att + (size_t)b * NE * FD, s_a, tid);

  for (int nt4 = 0; nt4 < 4; ++nt4) {
    no_hoist();                      // force per-job ds_load of A tiles (no spill)
    int c = (wid * 4 + nt4) * 16 + n;
    const __bf16* WT = wmT + (size_t)c * FD;
    float bias = bmlp[c];
    v8f acc[8];
#pragma unroll
    for (int m = 0; m < 8; ++m)
#pragma unroll
      for (int r = 0; r < 8; ++r) acc[m][r] = bias;

    for (int ks = 0; ks < 16; ++ks) {
      v16bf bm = *(const v16bf*)(WT + ks * 32 + kh * 16);
#pragma unroll
      for (int m = 0; m < 8; ++m) {
        v16bf am = load_a_tile(s_a, LDA, m * 16, ks * 32, n, kh);
        acc[m] = wmma_bf16(am, bm, acc[m]);
      }
    }
#pragma unroll
    for (int m = 0; m < 8; ++m)
#pragma unroll
      for (int r = 0; r < 8; ++r) {
        size_t idx = (size_t)(m * 16 + r + 8 * kh) * FD + c;
        yr[idx] = acc[m][r] + xr[idx];
      }
  }
}

// ---- K4: LayerNorm-post + masked mean pool -> out [b][512] f32 ----
__global__ void __launch_bounds__(256)
k_pool(const float* __restrict__ y, const float* __restrict__ mask,
       const float* __restrict__ g, const float* __restrict__ bp,
       float* __restrict__ out) {
  __shared__ float s_acc[8][FD];
  __shared__ float s_ms[8];
  int b = blockIdx.x;
  int wid = threadIdx.x >> 5, lane = threadIdx.x & 31;

  v4f g4[4], b4[4];
#pragma unroll
  for (int j = 0; j < 4; ++j) {
    int c0 = j * 128 + lane * 4;
    g4[j] = *(const v4f*)(g + c0);
    b4[j] = *(const v4f*)(bp + c0);
  }
  v4f acc[4] = {};
  float ms = 0.f;
  for (int ti = 0; ti < 16; ++ti) {
    int tok = wid * 16 + ti;
    float mr = mask[(size_t)b * NE + tok];
    const float* yrow = y + ((size_t)b * NE + tok) * FD;
    v4f u[4];
#pragma unroll
    for (int j = 0; j < 4; ++j) u[j] = *(const v4f*)(yrow + j * 128 + lane * 4);
    float s = 0.f;
#pragma unroll
    for (int j = 0; j < 4; ++j)
#pragma unroll
      for (int i = 0; i < 4; ++i) s += u[j][i];
    float mu = waveRedSum(s) * (1.f / FD);
    float vs = 0.f;
#pragma unroll
    for (int j = 0; j < 4; ++j)
#pragma unroll
      for (int i = 0; i < 4; ++i) { float d = u[j][i] - mu; vs += d * d; }
    float rs = rsqrtf(waveRedSum(vs) * (1.f / FD) + 1e-3f);
#pragma unroll
    for (int j = 0; j < 4; ++j)
#pragma unroll
      for (int i = 0; i < 4; ++i)
        acc[j][i] += ((u[j][i] - mu) * rs * g4[j][i] + b4[j][i]) * mr;
    ms += mr;
  }
#pragma unroll
  for (int j = 0; j < 4; ++j) *(v4f*)&s_acc[wid][j * 128 + lane * 4] = acc[j];
  if (lane == 0) s_ms[wid] = ms;
  __syncthreads();

  float denom = 1e-5f;
#pragma unroll
  for (int w = 0; w < 8; ++w) denom += s_ms[w];
#pragma unroll
  for (int e = 0; e < 2; ++e) {
    int c = threadIdx.x * 2 + e;
    float t = 0.f;
#pragma unroll
    for (int w = 0; w < 8; ++w) t += s_acc[w][c];
    out[(size_t)b * FD + c] = t / denom;
  }
}

extern "C" void kernel_launch(void* const* d_in, const int* in_sizes, int n_in,
                              void* d_out, int out_size, void* d_ws, size_t ws_size,
                              hipStream_t stream) {
  const float* x      = (const float*)d_in[0];
  const float* mask   = (const float*)d_in[1];
  const float* Wqk    = (const float*)d_in[2];
  const float* bqk    = (const float*)d_in[3];
  const float* Wv     = (const float*)d_in[4];
  const float* bv     = (const float*)d_in[5];
  const float* Wmlp   = (const float*)d_in[6];
  const float* bmlp   = (const float*)d_in[7];
  const float* g_pre  = (const float*)d_in[8];
  const float* b_pre  = (const float*)d_in[9];
  const float* g_post = (const float*)d_in[10];
  const float* b_post = (const float*)d_in[11];

  char* ws = (char*)d_ws;
  __bf16* inp  = (__bf16*)(ws + OFF_INP);
  __bf16* qb   = (__bf16*)(ws + OFF_Q);
  __bf16* kb   = (__bf16*)(ws + OFF_K);
  __bf16* vb   = (__bf16*)(ws + OFF_V);
  __bf16* wqkT = (__bf16*)(ws + OFF_WQKT);
  __bf16* wvT  = (__bf16*)(ws + OFF_WVT);
  __bf16* wmT  = (__bf16*)(ws + OFF_WMT);
  __bf16* attb = (__bf16*)(ws + OFF_ATT);
  float*  yb   = (float*)(ws + OFF_Y);

  k_transpose_cvt<<<(1024 * FD + 255) / 256, 256, 0, stream>>>(Wqk, wqkT, 1024);
  k_transpose_cvt<<<(512 * FD + 255) / 256, 256, 0, stream>>>(Wv, wvT, 512);
  k_transpose_cvt<<<(512 * FD + 255) / 256, 256, 0, stream>>>(Wmlp, wmT, 512);
  k_ln_pre<<<(BSZ * NE) / 8, 256, 0, stream>>>(x, g_pre, b_pre, inp);
  k_qkv<<<BSZ, 256, SMEM_ACT, stream>>>(inp, wqkT, wvT, bqk, bv, qb, kb, vb);
  k_attn<<<dim3(NH, BSZ), 256, 0, stream>>>(qb, kb, vb, mask, attb);
  k_outproj<<<BSZ, 256, SMEM_ACT, stream>>>(attb, wmT, bmlp, x, yb);
  k_pool<<<BSZ, 256, 0, stream>>>(yb, mask, g_post, b_post, (float*)d_out);
}
